// Monster_90348932039371
// MI455X (gfx1250) — compile-verified, matrix operand order checked by reading, not run on hardware
//
#include <hip/hip_runtime.h>

typedef __attribute__((ext_vector_type(16))) _Float16 v16h;
typedef __attribute__((ext_vector_type(8)))  _Float16 v8h;
typedef __attribute__((ext_vector_type(8)))  float    v8f;

// ---------------------------------------------------------------------------
// Pack torch-layout f32 weights (Cout, CinTot, KH, KW) -> f16 GEMM matrix
// [CoutPad][Kpad] selecting channel slice [ci0, ci0+CinSub), zero padded.
// ---------------------------------------------------------------------------
__global__ __launch_bounds__(256) void pack_weights_f16(
    const float* __restrict__ w, _Float16* __restrict__ dst,
    int Cout, int CinTot, int ci0, int CinSub, int KHW, int Kpad, int total)
{
    int idx = blockIdx.x * 256 + threadIdx.x;
    if (idx >= total) return;
    int m = idx / Kpad;
    int k = idx - m * Kpad;
    float v = 0.f;
    if (m < Cout && k < CinSub * KHW) {
        int ci = ci0 + k / KHW;
        int r  = k % KHW;
        v = w[((size_t)m * CinTot + ci) * KHW + r];
    }
    dst[idx] = (_Float16)v;
}

// ---------------------------------------------------------------------------
// Implicit-GEMM conv (stride 1, symmetric pad) on the WMMA pipe.
//   out[m][n] = sum_k Wf16[m][k] * im2col[k][n]   (f32 accumulate)
//   n = b*H*W + oh*W + ow ;  k = ci*KHW + kh*KW + kw
//
// Block = 128 threads = 4 wave32.
//   blockIdx.x           -> pair of N tiles (32 columns, shared by all waves)
//   blockIdx.y*4 + wave  -> M tile
// Per 32-K step the block gathers TWO 32x16 f16 B fragments into LDS
// (double-buffered => one barrier per K step); each wave loads one A fragment
// (two 16-B loads of packed weights) and issues two v_wmma_f32_16x16x32_f16,
// one per N tile.  Next step's gather loads are issued before the WMMAs so
// global latency hides under the matrix ops.
//
// Gather ownership: thread t fills fragment half-elements [8t, 8t+8), i.e.
// a single column n and EIGHT CONSECUTIVE k values -> one aligned
// ds_store_b128 per thread per step, one n-decomposition per thread, and
// contiguous global addresses inside a kernel row.  Branchless: clamped
// always-valid address + select-to-zero.
// ---------------------------------------------------------------------------
template <int KHW, int KW, bool ACC>
__global__ __launch_bounds__(128) void conv_implicit_gemm_wmma(
    const float*    __restrict__ x,     // NCHW f32 activations
    const _Float16* __restrict__ wh,    // [CoutPad][Kpad] f16 weights
    const float*    __restrict__ bias,  // [Cout] or null (ignored when ACC)
    float*          __restrict__ out,   // [Cout][Ntot] f32
    int Cin, int H, int W, int pad,
    int Kreal, int Kpad, int Cout, int CoutPad,
    int Ntot, int HW)
{
    const int tid  = threadIdx.x;
    const int wave = tid >> 5;
    const int lane = tid & 31;
    const int tilesM = CoutPad >> 4;
    const int nBase  = blockIdx.x * 32;
    const int tileM  = blockIdx.y * 4 + wave;
    const bool waveActive = (tileM < tilesM);   // wave-uniform: EXEC stays full

    __shared__ _Float16 ldsB[2][1024];          // [buf][2 fragments of 32x16]

    // --- gather plan: thread t owns fragment halves [8t, 8t+8) ------------
    //   idx = 8t+j : tile = t>>6 ; srcLane = (t>>1)&31 ; e = 8*(t&1)+j
    //   k offset   = koff(srcLane) + (t&1)*16 + j   (8 consecutive k)
    const int gTile  = tid >> 6;
    const int gSLane = (tid >> 1) & 31;
    const int kbase  = ((gSLane >= 16) ? 8 : 0) + (tid & 1) * 16;
    const int gN     = nBase + (gTile << 4) + (gSLane & 15);
    const bool gVal  = (gN < Ntot);
    int gOh = 0, gOw = 0, gXB = 0;
    if (gVal) {
        int bb = gN / HW;
        int r  = gN - bb * HW;
        gOh = r / W; gOw = r - gOh * W;
        gXB = bb * Cin * HW;                    // fits in int for these sizes
    }

    // --- A fragment addressing -------------------------------------------
    const int mRow  = tileM * 16 + (lane & 15);
    const int koffA = (lane >= 16) ? 8 : 0;
    const _Float16* wrow = wh + (size_t)mRow * Kpad;

    v8f c0, c1;
    #pragma unroll
    for (int i = 0; i < 8; ++i) { c0[i] = 0.f; c1[i] = 0.f; }

    // --- prologue: gather K-step 0 into buffer 0 --------------------------
    {
        v8h hv;
        #pragma unroll
        for (int j = 0; j < 8; ++j) {
            int kk = kbase + j;
            bool inK = (kk < Kreal);
            int kkc = inK ? kk : 0;
            int ci = kkc / KHW;  int r  = kkc - ci * KHW;
            int kh = r / KW;     int kw = r - kh * KW;
            int ih = gOh + kh - pad, iw = gOw + kw - pad;
            bool ok = gVal && inK &&
                      (unsigned)ih < (unsigned)H && (unsigned)iw < (unsigned)W;
            float v = x[(size_t)(gXB + ci * HW + (ok ? ih : 0) * W + (ok ? iw : 0))];
            hv[j] = (_Float16)(ok ? v : 0.f);
        }
        *(v8h*)(&ldsB[0][tid * 8]) = hv;
    }
    __syncthreads();

    int ibuf = 0;
    for (int k0 = 0; k0 < Kpad; k0 += 32, ibuf ^= 1) {
        const bool more = (k0 + 32 < Kpad);     // uniform

        // ---- issue next step's gather loads first (latency hiding) -------
        float vals[8];
        bool  oks[8];
        if (more) {
            #pragma unroll
            for (int j = 0; j < 8; ++j) {
                int kk = k0 + 32 + kbase + j;
                bool inK = (kk < Kreal);
                int kkc = inK ? kk : 0;
                int ci = kkc / KHW;  int r  = kkc - ci * KHW;
                int kh = r / KW;     int kw = r - kh * KW;
                int ih = gOh + kh - pad, iw = gOw + kw - pad;
                bool ok = gVal && inK &&
                          (unsigned)ih < (unsigned)H && (unsigned)iw < (unsigned)W;
                oks[j]  = ok;
                vals[j] = x[(size_t)(gXB + ci * HW +
                                     (ok ? ih : 0) * W + (ok ? iw : 0))];
            }
        }

        // ---- compute from current buffer ---------------------------------
        if (waveActive) {
            v8h alo = *(const v8h*)(wrow + k0 + koffA);
            v8h ahi = *(const v8h*)(wrow + k0 + 16 + koffA);
            v16h a;
            #pragma unroll
            for (int i = 0; i < 8; ++i) { a[i] = alo[i]; a[i + 8] = ahi[i]; }

            const v8h* bp0 = (const v8h*)(&ldsB[ibuf][0]   + lane * 16);
            const v8h* bp1 = (const v8h*)(&ldsB[ibuf][512] + lane * 16);
            v8h b0l = bp0[0], b0h = bp0[1];
            v8h b1l = bp1[0], b1h = bp1[1];
            v16h b0, b1;
            #pragma unroll
            for (int i = 0; i < 8; ++i) {
                b0[i] = b0l[i]; b0[i + 8] = b0h[i];
                b1[i] = b1l[i]; b1[i + 8] = b1h[i];
            }
            c0 = __builtin_amdgcn_wmma_f32_16x16x32_f16(
                false, a, false, b0, (short)0, c0, false, false);
            c1 = __builtin_amdgcn_wmma_f32_16x16x32_f16(
                false, a, false, b1, (short)0, c1, false, false);
        }

        // ---- write next fragments to the other buffer (1x ds_store_b128) -
        if (more) {
            v8h hv;
            #pragma unroll
            for (int j = 0; j < 8; ++j)
                hv[j] = (_Float16)(oks[j] ? vals[j] : 0.f);
            *(v8h*)(&ldsB[ibuf ^ 1][tid * 8]) = hv;
        }
        __syncthreads();
    }

    // --- epilogue: C layout N=lane%16, M = vgpr + 8*(lane>=16) -------------
    if (waveActive) {
        const int mBase = tileM * 16 + ((lane >= 16) ? 8 : 0);
        float badd[8];
        #pragma unroll
        for (int g = 0; g < 8; ++g) {
            int m = mBase + g;
            badd[g] = (!ACC && bias != nullptr && m < Cout) ? bias[m] : 0.f;
        }
        #pragma unroll
        for (int t = 0; t < 2; ++t) {
            int n = nBase + t * 16 + (lane & 15);
            if (n < Ntot) {
                #pragma unroll
                for (int g = 0; g < 8; ++g) {
                    int m = mBase + g;
                    if (m < Cout) {
                        size_t o = (size_t)m * Ntot + n;
                        float v = (t ? c1[g] : c0[g]) + badd[g];
                        if (ACC) v += out[o];
                        out[o] = v;
                    }
                }
            }
        }
    }
}

// ---------------------------------------------------------------------------
// InstanceNorm (biased var over HW) + ReLU + optional 1x1-residual add,
// reading the GEMM layout [Cout][B*HW] and writing NCHW into d_out.
// grid = (B, Cout), block = 256.
// ---------------------------------------------------------------------------
__global__ __launch_bounds__(256) void inorm_relu_residual(
    const float* __restrict__ conv, const float* __restrict__ res,
    float* __restrict__ out, int Cout, int HW, int Ntot)
{
    const int b = blockIdx.x, c = blockIdx.y, tid = threadIdx.x;
    const float* row = conv + (size_t)c * Ntot + (size_t)b * HW;

    float s = 0.f, s2 = 0.f;
    for (int i = tid; i < HW; i += 256) { float v = row[i]; s += v; s2 += v * v; }

    __shared__ float ss[256], sq[256];
    ss[tid] = s; sq[tid] = s2;
    __syncthreads();
    for (int st = 128; st > 0; st >>= 1) {
        if (tid < st) { ss[tid] += ss[tid + st]; sq[tid] += sq[tid + st]; }
        __syncthreads();
    }
    float mean = ss[0] / (float)HW;
    float var  = sq[0] / (float)HW - mean * mean;
    float inv  = rsqrtf(var + 1e-5f);

    float* orow = out + ((size_t)b * Cout + c) * HW;
    if (res != nullptr) {
        const float* rrow = res + (size_t)c * Ntot + (size_t)b * HW;
        for (int i = tid; i < HW; i += 256) {
            float v = (row[i] - mean) * inv;
            v = v > 0.f ? v : 0.f;
            orow[i] = v + rrow[i];
        }
    } else {
        for (int i = tid; i < HW; i += 256) {
            float v = (row[i] - mean) * inv;
            orow[i] = v > 0.f ? v : 0.f;
        }
    }
}

// ---------------------------------------------------------------------------
// ConvTranspose2d(k=3, stride=2, pad=1, output_pad=1, bias=False), direct
// gather form:  oh = 2*ih - 1 + kh  =>  even oh: (kh=1, ih=oh/2);
// odd oh: (kh=0, ih=(oh+1)/2 if <H) and (kh=2, ih=(oh-1)/2).  Same for W.
// Weight layout torch (Cin, Cout, 3, 3); per-co slice staged in LDS.
// grid = (B*Cout, spatial chunks), block = 256.
// ---------------------------------------------------------------------------
__global__ __launch_bounds__(256) void convT2x_kernel(
    const float* __restrict__ x, const float* __restrict__ wt,
    float* __restrict__ out, int Cin, int Cout, int H, int W)
{
    const int bco = blockIdx.x;
    const int b = bco / Cout, co = bco - b * Cout;

    __shared__ float wlds[1728];                        // up to 192*9
    for (int i = threadIdx.x; i < Cin * 9; i += 256) {
        int ci = i / 9, r = i - ci * 9;
        wlds[i] = wt[((size_t)ci * Cout + co) * 9 + r];
    }
    __syncthreads();

    const int W2 = 2 * W, HW = H * W, total = 4 * HW;
    const float* xb = x + (size_t)b * Cin * HW;

    for (int idx = blockIdx.y * 256 + threadIdx.x; idx < total;
         idx += gridDim.y * 256) {
        int oh = idx / W2, ow = idx - oh * W2;

        int khs[2], ihs[2], nv = 0;
        if (oh & 1) {
            int ih0 = (oh + 1) >> 1;
            if (ih0 < H) { khs[nv] = 0; ihs[nv] = ih0; ++nv; }
            khs[nv] = 2; ihs[nv] = (oh - 1) >> 1; ++nv;
        } else { khs[0] = 1; ihs[0] = oh >> 1; nv = 1; }

        int kws[2], iws[2], nh = 0;
        if (ow & 1) {
            int iw0 = (ow + 1) >> 1;
            if (iw0 < W) { kws[nh] = 0; iws[nh] = iw0; ++nh; }
            kws[nh] = 2; iws[nh] = (ow - 1) >> 1; ++nh;
        } else { kws[0] = 1; iws[0] = ow >> 1; nh = 1; }

        float accv = 0.f;
        for (int ci = 0; ci < Cin; ++ci) {
            const float* xr = xb + (size_t)ci * HW;
            const float* wr = wlds + ci * 9;
            for (int a = 0; a < nv; ++a)
                for (int c2 = 0; c2 < nh; ++c2)
                    accv += xr[ihs[a] * W + iws[c2]] * wr[khs[a] * 3 + kws[c2]];
        }
        out[(size_t)bco * total + idx] = accv;
    }
}

// ---------------------------------------------------------------------------
// Host orchestration
// ---------------------------------------------------------------------------
extern "C" void kernel_launch(void* const* d_in, const int* in_sizes, int n_in,
                              void* d_out, int out_size, void* d_ws, size_t ws_size,
                              hipStream_t stream)
{
    (void)in_sizes; (void)n_in; (void)out_size; (void)ws_size;

    const float* f4   = (const float*)d_in[0];
    const float* f8   = (const float*)d_in[1];
    const float* f16i = (const float*)d_in[2];
    const float* f32i = (const float*)d_in[3];
    const float* w32  = (const float*)d_in[4];
    const float* b32  = (const float*)d_in[5];
    const float* wt32 = (const float*)d_in[6];
    const float* w16  = (const float*)d_in[7];
    const float* b16  = (const float*)d_in[8];
    const float* wr16 = (const float*)d_in[9];
    const float* br16 = (const float*)d_in[10];
    const float* wt16 = (const float*)d_in[11];
    const float* w8   = (const float*)d_in[12];
    const float* b8   = (const float*)d_in[13];
    const float* wr8  = (const float*)d_in[14];
    const float* br8  = (const float*)d_in[15];
    const float* wt8  = (const float*)d_in[16];
    const float* w4   = (const float*)d_in[17];
    const float* b4   = (const float*)d_in[18];
    const float* wr4  = (const float*)d_in[19];
    const float* br4  = (const float*)d_in[20];

    // d_out: feat4, feat8, feat16, feat32 concatenated flat (NCHW each)
    float* out    = (float*)d_out;
    float* feat4  = out;
    float* feat8  = feat4  + (size_t)2 * 48  * 160 * 240;
    float* feat16 = feat8  + (size_t)2 * 64  * 80  * 120;
    float* feat32 = feat16 + (size_t)2 * 192 * 40  * 60;

    // workspace bump allocator (reused buffers, ~49 MB total)
    char* p = (char*)d_ws;
    auto alloc = [&p](size_t bytes) -> char* {
        char* r = p; p += (bytes + 255) & ~(size_t)255; return r;
    };
    float*    upBuf = (float*)alloc((size_t)3686400 * sizeof(float));
    float*    convB = (float*)alloc((size_t)3686400 * sizeof(float));
    float*    resB  = (float*)alloc((size_t)3686400 * sizeof(float));
    _Float16* wpool = (_Float16*)alloc((size_t)2500000 * 2);

    auto packLaunch = [&](const float* w, _Float16* dst, int Cout, int CinTot,
                          int ci0, int CinSub, int KHW, int Kpad, int CoutPad) {
        int total = CoutPad * Kpad;
        pack_weights_f16<<<(total + 255) / 256, 256, 0, stream>>>(
            w, dst, Cout, CinTot, ci0, CinSub, KHW, Kpad, total);
    };

    // ---------------- level 32: conv3x3(256->160) + IN + ReLU, convT ->192
    {
        const int Cout = 160, Cin = 256, H = 20, W = 30, HW = 600, Ntot = 1200;
        const int Kreal = Cin * 9, Kpad = 2304, CoutPad = 160;
        _Float16* Wh = wpool;
        packLaunch(w32, Wh, Cout, Cin, 0, Cin, 9, Kpad, CoutPad);
        dim3 g((Ntot + 31) / 32, (CoutPad / 16 + 3) / 4);
        conv_implicit_gemm_wmma<9, 3, false><<<g, 128, 0, stream>>>(
            f32i, Wh, b32, convB, Cin, H, W, 1, Kreal, Kpad, Cout, CoutPad, Ntot, HW);
        inorm_relu_residual<<<dim3(2, Cout), 256, 0, stream>>>(
            convB, nullptr, feat32, Cout, HW, Ntot);
        convT2x_kernel<<<dim3(2 * 192, (4 * HW + 255) / 256), 256, 0, stream>>>(
            feat32, wt32, upBuf, 160, 192, H, W);         // up32: [2,192,40,60]
    }

    // ---------------- level 16: conv5x5(448->192) split + IN/ReLU + res1x1
    {
        const int Cout = 192, H = 40, W = 60, HW = 2400, Ntot = 4800, CoutPad = 192;
        _Float16* Wa = wpool;                    // 192*6400
        _Float16* Wb = Wa + (size_t)192 * 6400;  // 192*4800
        _Float16* Wr = Wb + (size_t)192 * 4800;  // 192*256
        packLaunch(w16,  Wa, Cout, 448, 0,   256, 25, 6400, CoutPad);
        packLaunch(w16,  Wb, Cout, 448, 256, 192, 25, 4800, CoutPad);
        packLaunch(wr16, Wr, Cout, 256, 0,   256, 1,  256,  CoutPad);
        dim3 g((Ntot + 31) / 32, (CoutPad / 16 + 3) / 4);
        conv_implicit_gemm_wmma<25, 5, false><<<g, 128, 0, stream>>>(
            f16i, Wa, b16, convB, 256, H, W, 2, 256 * 25, 6400, Cout, CoutPad, Ntot, HW);
        conv_implicit_gemm_wmma<25, 5, true><<<g, 128, 0, stream>>>(
            upBuf, Wb, nullptr, convB, 192, H, W, 2, 192 * 25, 4800, Cout, CoutPad, Ntot, HW);
        conv_implicit_gemm_wmma<1, 1, false><<<g, 128, 0, stream>>>(
            f16i, Wr, br16, resB, 256, H, W, 0, 256, 256, Cout, CoutPad, Ntot, HW);
        inorm_relu_residual<<<dim3(2, Cout), 256, 0, stream>>>(
            convB, resB, feat16, Cout, HW, Ntot);
        convT2x_kernel<<<dim3(2 * 64, (4 * HW + 255) / 256), 256, 0, stream>>>(
            feat16, wt16, upBuf, 192, 64, H, W);          // up16: [2,64,80,120]
    }

    // ---------------- level 8: conv5x5(320->64) split + IN/ReLU + res1x1
    {
        const int Cout = 64, H = 80, W = 120, HW = 9600, Ntot = 19200, CoutPad = 64;
        _Float16* Wa = wpool;                    // 64*6400
        _Float16* Wb = Wa + (size_t)64 * 6400;   // 64*1600
        _Float16* Wr = Wb + (size_t)64 * 1600;   // 64*256
        packLaunch(w8,  Wa, Cout, 320, 0,   256, 25, 6400, CoutPad);
        packLaunch(w8,  Wb, Cout, 320, 256, 64,  25, 1600, CoutPad);
        packLaunch(wr8, Wr, Cout, 256, 0,   256, 1,  256,  CoutPad);
        dim3 g((Ntot + 31) / 32, (CoutPad / 16 + 3) / 4);
        conv_implicit_gemm_wmma<25, 5, false><<<g, 128, 0, stream>>>(
            f8, Wa, b8, convB, 256, H, W, 2, 256 * 25, 6400, Cout, CoutPad, Ntot, HW);
        conv_implicit_gemm_wmma<25, 5, true><<<g, 128, 0, stream>>>(
            upBuf, Wb, nullptr, convB, 64, H, W, 2, 64 * 25, 1600, Cout, CoutPad, Ntot, HW);
        conv_implicit_gemm_wmma<1, 1, false><<<g, 128, 0, stream>>>(
            f8, Wr, br8, resB, 256, H, W, 0, 256, 256, Cout, CoutPad, Ntot, HW);
        inorm_relu_residual<<<dim3(2, Cout), 256, 0, stream>>>(
            convB, resB, feat8, Cout, HW, Ntot);
        convT2x_kernel<<<dim3(2 * 48, (4 * HW + 255) / 256), 256, 0, stream>>>(
            feat8, wt8, upBuf, 64, 48, H, W);             // up8: [2,48,160,240]
    }

    // ---------------- level 4: conv5x5(304->48) split + IN/ReLU + res1x1
    {
        const int Cout = 48, H = 160, W = 240, HW = 38400, Ntot = 76800, CoutPad = 48;
        _Float16* Wa = wpool;                    // 48*6400
        _Float16* Wb = Wa + (size_t)48 * 6400;   // 48*1216 (Kreal=1200)
        _Float16* Wr = Wb + (size_t)48 * 1216;   // 48*256
        packLaunch(w4,  Wa, Cout, 304, 0,   256, 25, 6400, CoutPad);
        packLaunch(w4,  Wb, Cout, 304, 256, 48,  25, 1216, CoutPad);
        packLaunch(wr4, Wr, Cout, 256, 0,   256, 1,  256,  CoutPad);
        dim3 g((Ntot + 31) / 32, (CoutPad / 16 + 3) / 4);
        conv_implicit_gemm_wmma<25, 5, false><<<g, 128, 0, stream>>>(
            f4, Wa, b4, convB, 256, H, W, 2, 256 * 25, 6400, Cout, CoutPad, Ntot, HW);
        conv_implicit_gemm_wmma<25, 5, true><<<g, 128, 0, stream>>>(
            upBuf, Wb, nullptr, convB, 48, H, W, 2, 48 * 25, 1216, Cout, CoutPad, Ntot, HW);
        conv_implicit_gemm_wmma<1, 1, false><<<g, 128, 0, stream>>>(
            f4, Wr, br4, resB, 256, H, W, 0, 256, 256, Cout, CoutPad, Ntot, HW);
        inorm_relu_residual<<<dim3(2, Cout), 256, 0, stream>>>(
            convB, resB, feat4, Cout, HW, Ntot);
    }
}
